// SetCriterion_Yolov3_57346403336687
// MI455X (gfx1250) — 8-bit matrix operand order repaired and checked by reading, compiled, not run on hardware
//
#include <hip/hip_runtime.h>
#include <hip/hip_bf16.h>

typedef __attribute__((ext_vector_type(8))) int v8i;

#define BB 16
#define NN 2048
#define CC 80
#define CONF_THR 0.5f
#define NMS_THR 0.5f

__global__ __launch_bounds__(256) void yolo_prep(
    const float* __restrict__ boxes, const float* __restrict__ obj,
    const float* __restrict__ logits, float4* __restrict__ xyxy_u,
    float* __restrict__ score, float* __restrict__ conf_u, int* __restrict__ cls_u) {
  const int gid = blockIdx.x * 256 + threadIdx.x;
  if (gid >= BB * NN) return;

  const float4 bx = ((const float4*)boxes)[gid];
  float4 o;
  o.x = bx.x - bx.z * 0.5f;
  o.y = bx.y - bx.w * 0.5f;
  o.z = bx.x + bx.z * 0.5f;
  o.w = bx.y + bx.w * 0.5f;

  const float4* lg = (const float4*)(logits + (size_t)gid * CC);
  float m = -__builtin_inff();
  int mi = 0;
#pragma unroll
  for (int c4 = 0; c4 < CC / 4; ++c4) {
    float4 v = lg[c4];
    if (v.x > m) { m = v.x; mi = c4 * 4 + 0; }
    if (v.y > m) { m = v.y; mi = c4 * 4 + 1; }
    if (v.z > m) { m = v.z; mi = c4 * 4 + 2; }
    if (v.w > m) { m = v.w; mi = c4 * 4 + 3; }
  }

  const float ob = obj[gid];
  const bool valid = ob > CONF_THR;
  xyxy_u[gid] = o;
  score[gid]  = valid ? ob : -__builtin_inff();
  conf_u[gid] = m;
  cls_u[gid]  = mi;
}

__global__ __launch_bounds__(1024) void yolo_sort(
    const float* __restrict__ score, const float4* __restrict__ xyxy_u,
    const float* __restrict__ conf_u, const int* __restrict__ cls_u,
    float4* __restrict__ xyxy_s, float* __restrict__ score_s,
    float* __restrict__ conf_s, int* __restrict__ cls_s) {
  const int b = blockIdx.x;
  const int t = threadIdx.x;
  const int base = b * NN;
  __shared__ float skey[NN];
  __shared__ int   sidx[NN];

  for (int l = t; l < NN; l += 1024) { skey[l] = score[base + l]; sidx[l] = l; }
  __syncthreads();

  for (int k = 2; k <= NN; k <<= 1) {
    for (int j = k >> 1; j > 0; j >>= 1) {
      for (int l = t; l < NN; l += 1024) {
        const int p = l ^ j;
        if (p > l) {
          const float ka = skey[l], kp = skey[p];
          const int ia = sidx[l], ip = sidx[p];
          const bool pb = (kp > ka) || ((kp == ka) && (ip < ia));
          const bool up = ((l & k) == 0);
          if (pb == up) { skey[l] = kp; skey[p] = ka; sidx[l] = ip; sidx[p] = ia; }
        }
      }
      __syncthreads();
    }
  }

  for (int l = t; l < NN; l += 1024) {
    const int idx = sidx[l];
    xyxy_s[base + l]  = xyxy_u[base + idx];
    score_s[base + l] = skey[l];
    conf_s[base + l]  = conf_u[base + idx];
    cls_s[base + l]   = cls_u[base + idx];
  }
}

__global__ __launch_bounds__(32) void yolo_mask(
    const float4* __restrict__ xyxy_s, const int* __restrict__ cls_s,
    unsigned* __restrict__ mask) {
  const int lane = threadIdx.x;
  const int id = blockIdx.x;
  const int colWord = id & 63;
  const int rowTile = (id >> 6) & 127;
  const int b = id >> 13;
  const int rowBase = rowTile << 4;
  const int colBase = colWord << 5;
  const int nb = b * NN;

  __shared__ float4 rbox[16];
  __shared__ float4 cbox[32];
  __shared__ int    rcls[16];
  __shared__ int    ccls[32];
  if (lane < 16) {
    rbox[lane] = xyxy_s[nb + rowBase + lane];
    rcls[lane] = cls_s[nb + rowBase + lane];
  }
  cbox[lane] = xyxy_s[nb + colBase + lane];
  ccls[lane] = cls_s[nb + colBase + lane];
  __syncthreads();

  const int h  = lane >> 4;
  const int lm = lane & 15;

  const int cr = rcls[lm];
  v8i a0 = {0, 0, 0, 0, 0, 0, 0, 0};
  v8i a1 = {0, 0, 0, 0, 0, 0, 0, 0};
#pragma unroll
  for (int v = 0; v < 8; ++v) {
    const int kb = ((v >> 1) << 4) + (h << 3) + ((v & 1) << 2);
    const int d0 = cr - kb;
    const int d1 = cr - 64 - kb;
    a0[v] = (d0 >= 0 && d0 < 4) ? (1 << (d0 * 8)) : 0;
    a1[v] = (d1 >= 0 && d1 < 4) ? (1 << (d1 * 8)) : 0;
  }

  unsigned bal[2][8];
#pragma unroll
  for (int s = 0; s < 2; ++s) {
    const int cc = ccls[s * 16 + lm];
    v8i b0 = {0, 0, 0, 0, 0, 0, 0, 0};
    v8i b1 = {0, 0, 0, 0, 0, 0, 0, 0};
#pragma unroll
    for (int v = 0; v < 8; ++v) {
      const int kb = ((v >> 1) << 4) + (h << 3) + ((v & 1) << 2);
      const int d0 = cc - kb;
      const int d1 = cc - 64 - kb;
      b0[v] = (d0 >= 0 && d0 < 4) ? (1 << (d0 * 8)) : 0;
      b1[v] = (d1 >= 0 && d1 < 4) ? (1 << (d1 * 8)) : 0;
    }

    v8i acc = {0, 0, 0, 0, 0, 0, 0, 0};
    acc = __builtin_amdgcn_wmma_i32_16x16x64_iu8(false, a0, false, b0, acc, false, false);
    acc = __builtin_amdgcn_wmma_i32_16x16x64_iu8(false, a1, false, b1, acc, false, false);

    const float4 cb = cbox[s * 16 + lm];
    const float areaC = (cb.z - cb.x + 1.0f) * (cb.w - cb.y + 1.0f);
    const int colG = colBase + s * 16 + lm;
#pragma unroll
    for (int v = 0; v < 8; ++v) {
      const int r = v + (h << 3);
      const float4 rb = rbox[r];
      const float areaR = (rb.z - rb.x + 1.0f) * (rb.w - rb.y + 1.0f);
      float iw = fminf(rb.z, cb.z) - fmaxf(rb.x, cb.x) + 1.0f;
      float ih = fminf(rb.w, cb.w) - fmaxf(rb.y, cb.y) + 1.0f;
      iw = fmaxf(iw, 0.0f);
      ih = fmaxf(ih, 0.0f);
      const float inter = iw * ih;
      const float iou = inter / (areaR + areaC - inter);
      const bool pred = (iou >= NMS_THR) && (acc[v] != 0) && (colG > rowBase + r);
      bal[s][v] = __builtin_amdgcn_ballot_w32(pred);
    }
  }

  if (lane == 0) {
#pragma unroll
    for (int v = 0; v < 8; ++v) {
      const unsigned w0 = (bal[0][v] & 0xFFFFu) | (bal[1][v] << 16);
      const unsigned w1 = (bal[0][v] >> 16) | (bal[1][v] & 0xFFFF0000u);
      mask[(size_t)(nb + rowBase + v) * 64 + colWord]     = w0;
      mask[(size_t)(nb + rowBase + v + 8) * 64 + colWord] = w1;
    }
  }
}

__global__ __launch_bounds__(32) void yolo_scan_out(
    const float4* __restrict__ xyxy_s, const float* __restrict__ score_s,
    const float* __restrict__ conf_s, const int* __restrict__ cls_s,
    const unsigned* __restrict__ mask, float* __restrict__ out) {
  const int b = blockIdx.x;
  const int lane = threadIdx.x;
  const int base = b * NN;
  __shared__ unsigned aliveW[64];

  for (int c = 0; c < 64; ++c) {
    const bool v = score_s[base + c * 32 + lane] > CONF_THR;
    const unsigned m = __builtin_amdgcn_ballot_w32(v);
    if (lane == 0) aliveW[c] = m;
  }
  __syncthreads();

  for (int i = 0; i < NN; ++i) {
    const unsigned w = aliveW[i >> 5];
    if ((w >> (i & 31)) & 1u) {
      const unsigned* mrow = mask + (size_t)(base + i) * 64;
      aliveW[lane]      &= ~mrow[lane];
      aliveW[lane + 32] &= ~mrow[lane + 32];
    }
    __syncthreads();
  }

  for (int c = 0; c < 64; ++c) {
    const int n = c * 32 + lane;
    const bool keep = (aliveW[c] >> lane) & 1u;
    const float4 bx = xyxy_s[base + n];
    float4 lo, hi;
    lo.x = (float)b;
    lo.y = keep ? bx.x : 0.0f;
    lo.z = keep ? bx.y : 0.0f;
    lo.w = keep ? bx.z : 0.0f;
    hi.x = keep ? bx.w : 0.0f;
    hi.y = keep ? score_s[base + n] : 0.0f;
    hi.z = keep ? conf_s[base + n] : 0.0f;
    hi.w = keep ? (float)cls_s[base + n] : 0.0f;
    float4* orow = (float4*)(out + (size_t)(base + n) * 8);
    orow[0] = lo;
    orow[1] = hi;
  }
}

extern "C" void kernel_launch(void* const* d_in, const int* in_sizes, int n_in,
                              void* d_out, int out_size, void* d_ws, size_t ws_size,
                              hipStream_t stream) {
  const float* boxes  = (const float*)d_in[0];
  const float* obj    = (const float*)d_in[1];
  const float* logits = (const float*)d_in[2];
  float* out = (float*)d_out;

  char* ws = (char*)d_ws;
  float4*   xyxy_u  = (float4*)(ws + (size_t)0);
  float4*   xyxy_s  = (float4*)(ws + ((size_t)512 << 10));
  float*    score   = (float*) (ws + ((size_t)1024 << 10));
  float*    score_s = (float*) (ws + ((size_t)1152 << 10));
  float*    conf_u  = (float*) (ws + ((size_t)1280 << 10));
  float*    conf_s  = (float*) (ws + ((size_t)1408 << 10));
  int*      cls_u   = (int*)   (ws + ((size_t)1536 << 10));
  int*      cls_s   = (int*)   (ws + ((size_t)1664 << 10));
  unsigned* mask    = (unsigned*)(ws + ((size_t)1792 << 10));

  yolo_prep<<<(BB * NN) / 256, 256, 0, stream>>>(boxes, obj, logits, xyxy_u,
                                                 score, conf_u, cls_u);
  yolo_sort<<<BB, 1024, 0, stream>>>(score, xyxy_u, conf_u, cls_u,
                                     xyxy_s, score_s, conf_s, cls_s);
  yolo_mask<<<BB * (NN / 16) * (NN / 32), 32, 0, stream>>>(xyxy_s, cls_s, mask);
  yolo_scan_out<<<BB, 32, 0, stream>>>(xyxy_s, score_s, conf_s, cls_s, mask, out);
}
